// RSA_42082089566857
// MI455X (gfx1250) — matrix operand order repaired and checked by reading, not compiled, not run on hardware
//
#include <hip/hip_runtime.h>

typedef __bf16 bf16;
typedef __attribute__((ext_vector_type(16))) __bf16 v16bf;
typedef __attribute__((ext_vector_type(8)))  float  v8f;
typedef unsigned int u32;
typedef __attribute__((ext_vector_type(4))) unsigned int u32x4;
typedef __attribute__((ext_vector_type(8))) int i32x8;
typedef __attribute__((ext_vector_type(4))) int i32x4;

union ABf { v16bf v; uint4 u4[2]; };
union BF4 { bf16 h[4]; unsigned long long u; };
union BF8 { uint4 q; bf16 h[8]; };

static __device__ __forceinline__ v8f zero8() {
    v8f v;
#pragma unroll
    for (int i = 0; i < 8; ++i) v[i] = 0.0f;
    return v;
}

static __device__ __forceinline__ v8f wmma_bf16(v16bf a, v16bf b, v8f c) {
    return __builtin_amdgcn_wmma_f32_16x16x32_bf16(false, a, false, b, (short)0, c, false, false);
}

// A fragment (16x32, bf16): lane (m + 16g) holds row m, K = {8g..8g+7, 8g+16..8g+23}
static __device__ __forceinline__ v16bf load_a_frag(const bf16* base, int ld) {
    const int lane = threadIdx.x & 31;
    const int g = lane >> 4, m = lane & 15;
    const bf16* p = base + (size_t)m * ld + 8 * g;
    ABf f;
    f.u4[0] = *(const uint4*)(p);
    f.u4[1] = *(const uint4*)(p + 16);
    return f.v;
}

// B fragment (32x16, bf16): lane (n + 16g) holds col n, K = {8g..8g+7, 8g+16..8g+23}
static __device__ __forceinline__ v16bf load_b_frag(const bf16* base, int ld) {
    const int lane = threadIdx.x & 31;
    const int g = lane >> 4, n = lane & 15;
    const bf16* p = base + (size_t)n * ld + 8 * g;
    ABf f;
    f.u4[0] = *(const uint4*)(p);
    f.u4[1] = *(const uint4*)(p + 16);
    return f.v;
}

// Tensor Data Mover: async 2D tile (bf16) global -> LDS (D# per CDNA5 ISA).
static __device__ __forceinline__ void tdm_load_2d(u32 lds_byte_addr, const void* gaddr,
                                                   u32 width_elems, u32 height_rows,
                                                   u32 row_stride_elems, u32 tensor_rows) {
    unsigned long long ga = (unsigned long long)(uintptr_t)gaddr;
    u32x4 g0;
    g0[0] = 1u;                                              // count=1 valid user descriptor
    g0[1] = lds_byte_addr;                                   // lds_addr [63:32]
    g0[2] = (u32)(ga & 0xffffffffu);                         // global_addr [95:64]
    g0[3] = (u32)((ga >> 32) & 0x01ffffffu) | 0x80000000u;   // addr[56:32] | type=2
    i32x8 g1;
    g1[0] = (int)(1u << 16);                                 // data_size=1 (2 bytes)
    g1[1] = (int)((width_elems & 0xffffu) << 16);            // tensor_dim0 lo
    g1[2] = (int)(((width_elems >> 16) & 0xffffu) | ((tensor_rows & 0xffffu) << 16));
    g1[3] = (int)(((tensor_rows >> 16) & 0xffffu) | ((width_elems & 0xffffu) << 16)); // tile_dim0
    g1[4] = (int)(height_rows & 0xffffu);                    // tile_dim1 (tile_dim2=0)
    g1[5] = (int)row_stride_elems;                           // tensor_dim0_stride lo
    g1[6] = 0;
    g1[7] = 0;
    i32x4 z4;
#pragma unroll
    for (int i = 0; i < 4; ++i) z4[i] = 0;
#if __clang_major__ >= 23
    i32x8 z8;
#pragma unroll
    for (int i = 0; i < 8; ++i) z8[i] = 0;
    __builtin_amdgcn_tensor_load_to_lds(g0, g1, z4, z4, z8, 0);
#else
    __builtin_amdgcn_tensor_load_to_lds(g0, g1, z4, z4, 0);
#endif
}

__global__ void cvt_kernel(const float* __restrict__ src, bf16* __restrict__ dst, int n) {
    int i = (blockIdx.x * blockDim.x + threadIdx.x) * 4;
    if (i < n) {
        float4 f = *(const float4*)(src + i);
        BF4 o;
        o.h[0] = (bf16)f.x; o.h[1] = (bf16)f.y; o.h[2] = (bf16)f.z; o.h[3] = (bf16)f.w;
        *(unsigned long long*)(dst + i) = o.u;
    }
}

// C = A[MxK] * W^T[KxN] + bias ; bf16 in, f32 acc; ping-pong pipelined (no reg rotation).
__global__ __launch_bounds__(256)
void gemm_kernel(const bf16* __restrict__ A, const bf16* __restrict__ W,
                 const float* __restrict__ bias, bf16* __restrict__ outb,
                 float* __restrict__ outf, int M, int N, int K) {
    const int tid = threadIdx.x, wv = tid >> 5, lane = tid & 31;
    const int g = lane >> 4, ln = lane & 15;
    const int m0 = blockIdx.y * 64 + (wv >> 1) * 16;
    const int n0 = blockIdx.x * 128 + (wv & 1) * 64;

    v8f acc[4];
#pragma unroll
    for (int c = 0; c < 4; ++c) acc[c] = zero8();

    const bf16* Abase = A + (size_t)m0 * K;
    const bf16* Wbase = W + (size_t)n0 * K;

    v16bf a[2], b[2][4];
    a[0] = load_a_frag(Abase, K);
#pragma unroll
    for (int c = 0; c < 4; ++c) b[0][c] = load_b_frag(Wbase + (size_t)c * 16 * K, K);

    for (int k0 = 0; k0 < K; k0 += 64) {
        // prefetch buffer 1 (k0+32) while computing on buffer 0 (k0)
        a[1] = load_a_frag(Abase + k0 + 32, K);
#pragma unroll
        for (int c = 0; c < 4; ++c) b[1][c] = load_b_frag(Wbase + (size_t)c * 16 * K + k0 + 32, K);
#pragma unroll
        for (int c = 0; c < 4; ++c) acc[c] = wmma_bf16(a[0], b[0][c], acc[c]);

        // prefetch buffer 0 (k0+64) while computing on buffer 1 (k0+32)
        const int kn = (k0 + 64 < K) ? (k0 + 64) : 0;  // last iter: dead prefetch
        a[0] = load_a_frag(Abase + kn, K);
#pragma unroll
        for (int c = 0; c < 4; ++c) b[0][c] = load_b_frag(Wbase + (size_t)c * 16 * K + kn, K);
#pragma unroll
        for (int c = 0; c < 4; ++c) acc[c] = wmma_bf16(a[1], b[1][c], acc[c]);
    }
#pragma unroll
    for (int r = 0; r < 8; ++r) {
        const int row = r + 8 * g;
#pragma unroll
        for (int c = 0; c < 4; ++c) {
            const int col = n0 + c * 16 + ln;
            float v = acc[c][r] + bias[col];
            size_t idx = (size_t)(m0 + row) * N + col;
            if (outb) outb[idx] = (bf16)v;
            else      outf[idx] = v;
        }
    }
}

// Fused flash-style attention + analytic positional blend, double-buffered LDS.
// Block: 256 thr (8 waves) = (batch, head, 128 q rows); wave owns 16 q rows.
__global__ __launch_bounds__(256)
void attn_kernel(const bf16* __restrict__ Qb, const bf16* __restrict__ Kb,
                 const bf16* __restrict__ Vb, bf16* __restrict__ Ab,
                 const float* __restrict__ mu, const float* __restrict__ lam,
                 const float* __restrict__ gc1, const float* __restrict__ gc2,
                 const float* __restrict__ tc1, const float* __restrict__ tc2) {
    __shared__ __align__(16) bf16 Ks[2][32 * 64];   // [buf][key][d]  (filled by TDM)
    __shared__ __align__(16) bf16 Vt[2][64 * 32];   // [buf][d][key]  (manual transpose)
    __shared__ __align__(16) bf16 Pb[8 * 16 * 32];  // per-wave softmax-P staging

    const int qt = blockIdx.x, h = blockIdx.y, bidx = blockIdx.z;
    const int tid = threadIdx.x, wv = tid >> 5, lane = tid & 31;
    const int g = lane >> 4, ln = lane & 15;

    float base, theta; int trig;
    if (h < 4)       { base = lam[h];      theta = 0.0f;        trig = 0; }
    else if (h < 10) { base = gc1[h - 4];  theta = tc1[h - 4];  trig = 1; }
    else             { base = gc2[h - 10]; theta = tc2[h - 10]; trig = 2; }
    const float l2b  = log2f(fabsf(base));
    const int   negb = (base < 0.0f);
    const float wbl  = 1.0f / (1.0f + __expf(-mu[0]));

    const size_t bt0 = (size_t)bidx * 2048;
    const int qbase = qt * 128 + wv * 16;
    const size_t qoff = (bt0 + qbase) * 1024 + (size_t)h * 64;
    const u32 ks_lds = (u32)(uintptr_t)(void*)&Ks[0][0];  // low 32 bits = LDS offset

    // stage tile k0s into LDS buffer `buf` (async TDM for K, manual transpose for V)
    auto stage = [&](int k0s, int buf) {
        if (wv == 0) {
            tdm_load_2d(ks_lds + (u32)buf * 4096u,
                        Kb + (bt0 + k0s) * 1024 + (size_t)h * 64,
                        64u, 32u, 1024u, (u32)(2048 - k0s));
        }
        const int kk = tid >> 3, dd0 = (tid & 7) * 8;
        BF8 vv;
        vv.q = *(const uint4*)(Vb + (bt0 + k0s + kk) * 1024 + (size_t)h * 64 + dd0);
#pragma unroll
        for (int i = 0; i < 8; ++i) Vt[buf][(dd0 + i) * 32 + kk] = vv.h[i];
    };

    v16bf aq0 = load_a_frag(Qb + qoff, 1024);
    v16bf aq1 = load_a_frag(Qb + qoff + 32, 1024);

    v8f o[4], pa[4];
#pragma unroll
    for (int c = 0; c < 4; ++c) { o[c] = zero8(); pa[c] = zero8(); }
    float mrow[8], lrow[8];
#pragma unroll
    for (int r = 0; r < 8; ++r) { mrow[r] = -1e30f; lrow[r] = 0.0f; }

    const int qi = qbase + ln;  // A-layout row for the positional fragment

    stage(0, 0);  // prologue

    for (int k0 = 0; k0 < 2048; k0 += 32) {
        const int cur = (k0 >> 5) & 1, nxt = cur ^ 1;
        if (wv == 0) __builtin_amdgcn_s_wait_tensorcnt(0);  // K tile (cur) done
        __syncthreads();  // cur buffer ready for all; prev reads of nxt done
        if (k0 + 32 < 2048) stage(k0 + 32, nxt);  // overlap next-tile DMA with compute

        const bf16* KsC = &Ks[cur][0];
        const bf16* VtC = &Vt[cur][0];

        // S = Q * K^T  (16 q rows x 32 keys)
        v8f s0 = zero8(), s1 = zero8();
        {
            v16bf b00 = load_b_frag(KsC + 0, 64);
            v16bf b01 = load_b_frag(KsC + 32, 64);
            v16bf b10 = load_b_frag(KsC + 16 * 64, 64);
            v16bf b11 = load_b_frag(KsC + 16 * 64 + 32, 64);
            s0 = wmma_bf16(aq0, b00, s0); s0 = wmma_bf16(aq1, b01, s0);
            s1 = wmma_bf16(aq0, b10, s1); s1 = wmma_bf16(aq1, b11, s1);
        }

        // online softmax (row = r + 8g, cols spread over 16-lane group)
        float rmx[8];
#pragma unroll
        for (int r = 0; r < 8; ++r) rmx[r] = fmaxf(s0[r], s1[r]);
#pragma unroll
        for (int msk = 1; msk < 16; msk <<= 1)
#pragma unroll
            for (int r = 0; r < 8; ++r) rmx[r] = fmaxf(rmx[r], __shfl_xor(rmx[r], msk, 16));
        float corr[8], rs[8];
#pragma unroll
        for (int r = 0; r < 8; ++r) {
            float mn = fmaxf(mrow[r], rmx[r]);
            corr[r] = __expf(mrow[r] - mn);
            mrow[r] = mn;
            s0[r] = __expf(s0[r] - mn);
            s1[r] = __expf(s1[r] - mn);
            rs[r] = s0[r] + s1[r];
        }
#pragma unroll
        for (int msk = 1; msk < 16; msk <<= 1)
#pragma unroll
            for (int r = 0; r < 8; ++r) rs[r] += __shfl_xor(rs[r], msk, 16);
#pragma unroll
        for (int r = 0; r < 8; ++r) lrow[r] = lrow[r] * corr[r] + rs[r];
#pragma unroll
        for (int c = 0; c < 4; ++c)
#pragma unroll
            for (int r = 0; r < 8; ++r) o[c][r] *= corr[r];

        // stage probabilities C-layout -> LDS (for A-layout reload)
#pragma unroll
        for (int r = 0; r < 8; ++r) {
            const int row = r + 8 * g;
            Pb[wv * 512 + row * 32 + ln]      = (bf16)s0[r];
            Pb[wv * 512 + row * 32 + 16 + ln] = (bf16)s1[r];
        }

        // analytic positional tile, generated directly in A-fragment layout
        ABf ap;
#pragma unroll
        for (int hh = 0; hh < 16; ++hh) {
            const int kk = (hh < 8 ? hh : hh + 8) + 8 * g;
            const float Lf = (float)(qi - (k0 + kk));
            float val = 0.0f;
            if (Lf >= 1.0f) {
                float L = fminf(Lf, 100.0f);
                float mag = exp2f(L * l2b);
                if (negb && (((int)L) & 1)) mag = -mag;
                if (trig == 1)      mag *= __sinf(theta * L);
                else if (trig == 2) mag *= __cosf(theta * L);
                val = mag;
            }
            ap.v[hh] = (bf16)val;
        }

        __syncthreads();  // Pb visible
        v16bf app = load_a_frag(Pb + wv * 512, 32);
#pragma unroll
        for (int c = 0; c < 4; ++c) {
            v16bf bvf = load_b_frag(VtC + c * 16 * 32, 32);
            o[c]  = wmma_bf16(app,  bvf, o[c]);   // att @ V (unnormalized)
            pa[c] = wmma_bf16(ap.v, bvf, pa[c]);  // P @ V
        }
    }

    // blend + store bf16 attention output
#pragma unroll
    for (int r = 0; r < 8; ++r) {
        const int row = r + 8 * g;
        const float invl = 1.0f / lrow[r];
        const size_t orow = (bt0 + qbase + row) * 1024 + (size_t)h * 64;
#pragma unroll
        for (int c = 0; c < 4; ++c) {
            float v = (1.0f - wbl) * o[c][r] * invl + wbl * pa[c][r];
            Ab[orow + c * 16 + ln] = (bf16)v;
        }
    }
}

extern "C" void kernel_launch(void* const* d_in, const int* in_sizes, int n_in,
                              void* d_out, int out_size, void* d_ws, size_t ws_size,
                              hipStream_t stream) {
    (void)in_sizes; (void)n_in; (void)out_size; (void)ws_size;
    const float* x   = (const float*)d_in[0];
    const float* mu  = (const float*)d_in[1];
    const float* lam = (const float*)d_in[2];
    const float* gc1 = (const float*)d_in[3];
    const float* gc2 = (const float*)d_in[4];
    const float* tc1 = (const float*)d_in[5];
    const float* tc2 = (const float*)d_in[6];
    const float* Wq  = (const float*)d_in[7];
    const float* bq  = (const float*)d_in[8];
    const float* Wk  = (const float*)d_in[9];
    const float* bk  = (const float*)d_in[10];
    const float* Wv  = (const float*)d_in[11];
    const float* bv  = (const float*)d_in[12];
    const float* Wf  = (const float*)d_in[13];
    const float* bfb = (const float*)d_in[14];

    const int XN = 4 * 2048 * 1024;   // 8388608
    const int WN = 1024 * 1024;       // 1048576

    bf16* xb  = (bf16*)d_ws;
    bf16* Wqb = xb  + XN;
    bf16* Wkb = Wqb + WN;
    bf16* Wvb = Wkb + WN;
    bf16* Wfb = Wvb + WN;
    bf16* Qb  = Wfb + WN;
    bf16* Kb  = Qb  + XN;
    bf16* Vb  = Kb  + XN;
    bf16* Ab  = Vb  + XN;

    cvt_kernel<<<(XN / 4 + 255) / 256, 256, 0, stream>>>(x,  xb,  XN);
    cvt_kernel<<<(WN / 4 + 255) / 256, 256, 0, stream>>>(Wq, Wqb, WN);
    cvt_kernel<<<(WN / 4 + 255) / 256, 256, 0, stream>>>(Wk, Wkb, WN);
    cvt_kernel<<<(WN / 4 + 255) / 256, 256, 0, stream>>>(Wv, Wvb, WN);
    cvt_kernel<<<(WN / 4 + 255) / 256, 256, 0, stream>>>(Wf, Wfb, WN);

    dim3 gg(1024 / 128, 8192 / 64);  // (8, 128)
    gemm_kernel<<<gg, 256, 0, stream>>>(xb, Wqb, bq, Qb, nullptr, 8192, 1024, 1024);
    gemm_kernel<<<gg, 256, 0, stream>>>(xb, Wkb, bk, Kb, nullptr, 8192, 1024, 1024);
    gemm_kernel<<<gg, 256, 0, stream>>>(xb, Wvb, bv, Vb, nullptr, 8192, 1024, 1024);

    dim3 ga(16, 16, 4);  // (qtile, head, batch)
    attn_kernel<<<ga, 256, 0, stream>>>(Qb, Kb, Vb, Ab, mu, lam, gc1, gc2, tc1, tc2);

    gemm_kernel<<<gg, 256, 0, stream>>>(Ab, Wfb, bfb, nullptr, (float*)d_out, 8192, 1024, 1024);
}
